// RNModule_27230092656812
// MI455X (gfx1250) — matrix operand order repaired and checked by reading, not compiled, not run on hardware
//
#include <hip/hip_runtime.h>
#include <hip/hip_fp16.h>
#include <hip/hip_bf16.h>

typedef __attribute__((ext_vector_type(16))) _Float16 v16h;
typedef __attribute__((ext_vector_type(8)))  float    v8f;
typedef int v4i_b128 __attribute__((vector_size(16)));

// Problem constants
static constexpr int BSZ  = 8;
static constexpr int CCH  = 128;
static constexpr int NPT  = 2048;
static constexpr int KNN  = 8;
static constexpr int PBN  = BSZ * NPT;   // 16384
static constexpr int PTOT = PBN * KNN;   // 131072

// ---------------------------------------------------------------------------
// CDNA5 async global->LDS copy (ASYNCcnt tracked), guarded for portability
// ---------------------------------------------------------------------------
#if defined(__has_builtin)
#if __has_builtin(__builtin_amdgcn_global_load_async_to_lds_b128)
#define HAVE_ASYNC_LDS 1
#endif
#endif
#ifndef HAVE_ASYNC_LDS
#define HAVE_ASYNC_LDS 0
#endif

__device__ __forceinline__ void cp16(const _Float16* __restrict__ g,
                                     _Float16* l) {
#if HAVE_ASYNC_LDS
    __builtin_amdgcn_global_load_async_to_lds_b128(
        (__attribute__((address_space(1))) v4i_b128*)(uintptr_t)g,
        (__attribute__((address_space(3))) v4i_b128*)(unsigned int)(uintptr_t)l,
        0, 0);
#else
    *(uint4*)l = *(const uint4*)g;
#endif
}

template <int N>
__device__ __forceinline__ void wait_async() {
#if HAVE_ASYNC_LDS
#if __has_builtin(__builtin_amdgcn_s_wait_asynccnt)
    __builtin_amdgcn_s_wait_asynccnt((short)N);
#else
    if constexpr (N == 0)
        asm volatile("s_wait_asynccnt 0x0" ::: "memory");
    else
        asm volatile("s_wait_asynccnt 0x4" ::: "memory");
#endif
#endif
}

// ---------------------------------------------------------------------------
// f32 -> f16 weight conversion (keeps [o][c] layout == k-contiguous B layout)
// ---------------------------------------------------------------------------
__global__ void cvt_f16_kernel(const float* __restrict__ in,
                               _Float16* __restrict__ out, int n) {
    int i = blockIdx.x * 256 + threadIdx.x;
    if (i < n) out[i] = (_Float16)in[i];
}

// ---------------------------------------------------------------------------
// feature (b, C, N) f32  ->  featT (b, N, C) f16   (32x32 LDS tiles)
// ---------------------------------------------------------------------------
__global__ __launch_bounds__(256) void transpose_kernel(
    const float* __restrict__ feature, _Float16* __restrict__ featT) {
    __shared__ float tile[32][33];
    const int b  = blockIdx.z;
    const int n0 = blockIdx.x * 32;
    const int c0 = blockIdx.y * 32;
    const float* src = feature + (size_t)b * CCH * NPT;
    _Float16*    dst = featT   + (size_t)b * NPT * CCH;
#pragma unroll
    for (int i = 0; i < 32; i += 8)
        tile[threadIdx.y + i][threadIdx.x] =
            src[(size_t)(c0 + threadIdx.y + i) * NPT + n0 + threadIdx.x];
    __syncthreads();
#pragma unroll
    for (int i = 0; i < 32; i += 8)
        dst[(size_t)(n0 + threadIdx.y + i) * CCH + c0 + threadIdx.x] =
            (_Float16)tile[threadIdx.x][threadIdx.y + i];
}

// ---------------------------------------------------------------------------
// KNN: per batch, top-9 smallest squared distances (stable ties), drop self
// ---------------------------------------------------------------------------
__global__ __launch_bounds__(256) void knn_kernel(
    const float* __restrict__ xyz, int* __restrict__ idxOut) {
    __shared__ float sx[NPT], sy[NPT], sz[NPT];
    const int b  = blockIdx.x >> 3;
    const int i0 = (blockIdx.x & 7) * 256;
    const float* base = xyz + (size_t)b * NPT * 3;
    for (int j = threadIdx.x; j < NPT; j += 256) {
        sx[j] = base[j * 3 + 0];
        sy[j] = base[j * 3 + 1];
        sz[j] = base[j * 3 + 2];
    }
    __syncthreads();
    const int i = i0 + threadIdx.x;
    const float xi = sx[i], yi = sy[i], zi = sz[i];
    float dd[9];
    int   ii[9];
#pragma unroll
    for (int t = 0; t < 9; ++t) { dd[t] = 3.4e38f; ii[t] = -1; }
    for (int j = 0; j < NPT; ++j) {
        const float dx = xi - sx[j], dy = yi - sy[j], dz = zi - sz[j];
        const float d  = dx * dx + dy * dy + dz * dz;
        if (d < dd[8]) {
            dd[8] = d; ii[8] = j;
#pragma unroll
            for (int t = 8; t > 0; --t) {
                if (dd[t] < dd[t - 1]) {
                    float td = dd[t]; dd[t] = dd[t - 1]; dd[t - 1] = td;
                    int   ti = ii[t]; ii[t] = ii[t - 1]; ii[t - 1] = ti;
                }
            }
        }
    }
    int* op = idxOut + ((size_t)b * NPT + i) * KNN;
#pragma unroll
    for (int k = 0; k < KNN; ++k) op[k] = ii[k + 1];
}

// ---------------------------------------------------------------------------
// rel[p=(b,n,k)][c] = featT[b,n,c] + featT[b,idx,c]   (f16)
// ---------------------------------------------------------------------------
__global__ __launch_bounds__(128) void gather_rel_kernel(
    const _Float16* __restrict__ featT, const int* __restrict__ idx,
    _Float16* __restrict__ rel) {
    const int bn = blockIdx.x;
    const int b  = bn >> 11;
    const int c  = threadIdx.x;
    const float self = (float)featT[(size_t)bn * CCH + c];
    const int* ip = idx + (size_t)bn * KNN;
    _Float16* op  = rel + ((size_t)bn * KNN) * CCH + c;
#pragma unroll
    for (int k = 0; k < KNN; ++k) {
        const int j = ip[k];
        const float v = self + (float)featT[((size_t)(b << 11) + j) * CCH + c];
        op[(size_t)k * CCH] = (_Float16)v;
    }
}

// ---------------------------------------------------------------------------
// WMMA GEMM: out[P][Cout] = A[P][Cin] * Bw[Cout][Cin]^T + bias
// MODE 0: store f16 rows     MODE 1: store f32 as (b, Cout, 2048) (rn_feature)
// block = 256 (8 waves), block tile 128x128, wave tile 64x32
// Double-buffered LDS, async global->LDS staging.
// ---------------------------------------------------------------------------
template <int MODE>
__global__ __launch_bounds__(256) void gemm_f16_kernel(
    const _Float16* __restrict__ A, const _Float16* __restrict__ Bw,
    const float* __restrict__ bias, _Float16* __restrict__ outH,
    float* __restrict__ outF, int Cin, int Cout) {
    __shared__ __align__(16) _Float16 As[2][128][40];
    __shared__ __align__(16) _Float16 Bs[2][128][40];

    const int tid  = threadIdx.x;
    const int lane = tid & 31;
    const int wave = tid >> 5;
    const int m0   = (wave & 1) * 64;
    const int n0   = (wave >> 1) * 32;
    const int l16  = lane & 15;
    const int hid  = lane >> 4;
    const int pBase = blockIdx.x * 128;
    const int nBase = blockIdx.y * 128;

    v8f acc[4][2];
#pragma unroll
    for (int mt = 0; mt < 4; ++mt)
#pragma unroll
        for (int nt = 0; nt < 2; ++nt)
#pragma unroll
            for (int r = 0; r < 8; ++r) acc[mt][nt][r] = 0.0f;

    const int srow = tid >> 1;
    const int scol = (tid & 1) << 4;
    const _Float16* gA = A  + (size_t)(pBase + srow) * Cin + scol;
    const _Float16* gB = Bw + (size_t)(nBase + srow) * Cin + scol;

    auto issue = [&](int kt, int buf) {
        cp16(gA + kt * 32,     &As[buf][srow][scol]);
        cp16(gA + kt * 32 + 8, &As[buf][srow][scol + 8]);
        cp16(gB + kt * 32,     &Bs[buf][srow][scol]);
        cp16(gB + kt * 32 + 8, &Bs[buf][srow][scol + 8]);
    };

    const int ntiles = Cin >> 5;
    issue(0, 0);

    for (int kt = 0; kt < ntiles; ++kt) {
        const int buf = kt & 1;
        if (kt + 1 < ntiles) {
            issue(kt + 1, buf ^ 1);   // prefetch next tile into other buffer
            wait_async<4>();          // tile kt complete (in-order)
        } else {
            wait_async<0>();
        }
        __syncthreads();

        union { uint4 q[2]; v16h v; } bf[2];
#pragma unroll
        for (int nt = 0; nt < 2; ++nt) {
            const _Float16* bp = &Bs[buf][n0 + nt * 16 + l16][hid * 8];
            bf[nt].q[0] = *(const uint4*)(bp);
            bf[nt].q[1] = *(const uint4*)(bp + 16);
        }
#pragma unroll
        for (int mt = 0; mt < 4; ++mt) {
            union { uint4 q[2]; v16h v; } af;
            const _Float16* ap = &As[buf][m0 + mt * 16 + l16][hid * 8];
            af.q[0] = *(const uint4*)(ap);
            af.q[1] = *(const uint4*)(ap + 16);
#pragma unroll
            for (int nt = 0; nt < 2; ++nt) {
                acc[mt][nt] = __builtin_amdgcn_wmma_f32_16x16x32_f16(
                    false, af.v, false, bf[nt].v, (short)0, acc[mt][nt],
                    false, false);
            }
        }
        __syncthreads();   // protect buf from being overwritten next iter
    }

    // epilogue: lane holds column gn, rows gm0..gm0+7
#pragma unroll
    for (int mt = 0; mt < 4; ++mt) {
        const int gm0 = pBase + m0 + mt * 16 + hid * 8;
#pragma unroll
        for (int nt = 0; nt < 2; ++nt) {
            const int gn = nBase + n0 + nt * 16 + l16;
            const float bb = bias[gn];
#pragma unroll
            for (int r = 0; r < 8; ++r) {
                const float v = acc[mt][nt][r] + bb;
                const int gm = gm0 + r;
                if (MODE == 1) {
                    const int b = gm >> 11, n = gm & 2047;
                    outF[(((size_t)b * 128 + gn) << 11) + n] = v;
                } else {
                    outH[(size_t)gm * Cout + gn] = (_Float16)v;
                }
            }
        }
    }
}

// ---------------------------------------------------------------------------
// per-channel sum / sumsq over rows (for BN stats)
// ---------------------------------------------------------------------------
__global__ __launch_bounds__(256) void stats_kernel(
    const _Float16* __restrict__ X, float* __restrict__ stats, int Cc,
    int rowsPerBlock) {
    const int t = threadIdx.x;
    const int c = t & (Cc - 1);
    const int rstep = 256 / Cc;
    const int rEnd  = blockIdx.x * rowsPerBlock + rowsPerBlock;
    float s = 0.f, ss = 0.f;
    for (int r = blockIdx.x * rowsPerBlock + (t / Cc); r < rEnd; r += rstep) {
        const float v = (float)X[(size_t)r * Cc + c];
        s += v; ss += v * v;
    }
    atomicAdd(&stats[c], s);
    atomicAdd(&stats[Cc + c], ss);
}

__global__ void finalize_bn_kernel(const float* __restrict__ stats,
                                   const float* __restrict__ g,
                                   const float* __restrict__ be,
                                   float* __restrict__ alpha,
                                   float* __restrict__ beta, int Cc,
                                   float invP) {
    const int c = blockIdx.x * blockDim.x + threadIdx.x;
    if (c < Cc) {
        const float m  = stats[c] * invP;
        const float v  = stats[Cc + c] * invP - m * m;
        const float rs = rsqrtf(v + 1e-5f);
        const float a  = g[c] * rs;
        alpha[c] = a;
        beta[c]  = be[c] - m * a;
    }
}

// ---------------------------------------------------------------------------
// hn[bn][o] = BN-affine( mean_k gu[(bn,k)][o] )    (f16)
// ---------------------------------------------------------------------------
__global__ __launch_bounds__(256) void reducek_kernel(
    const _Float16* __restrict__ gu, const float* __restrict__ alpha,
    const float* __restrict__ beta, _Float16* __restrict__ hn) {
    const int bn = blockIdx.x;
    const int o  = threadIdx.x;
    const _Float16* ip = gu + ((size_t)bn * KNN) * 256 + o;
    float s = 0.f;
#pragma unroll
    for (int k = 0; k < KNN; ++k) s += (float)ip[(size_t)k * 256];
    s = s * 0.125f * alpha[o] + beta[o];
    hn[(size_t)bn * 256 + o] = (_Float16)s;
}

// ---------------------------------------------------------------------------
// x = relu(y*alpha + beta)  elementwise over [P][128] f16 (vectorized 8-wide)
// ---------------------------------------------------------------------------
__global__ __launch_bounds__(256) void affine_relu_kernel(
    const _Float16* __restrict__ in, const float* __restrict__ alpha,
    const float* __restrict__ beta, _Float16* __restrict__ out) {
    __shared__ float sa[128], sb[128];
    const int t = threadIdx.x;
    if (t < 128) { sa[t] = alpha[t]; sb[t] = beta[t]; }
    __syncthreads();
    const int row = blockIdx.x * 16 + (t >> 4);
    const int c0  = (t & 15) * 8;
    const size_t o = (size_t)row * 128 + c0;
    union { uint4 q; _Float16 h[8]; } u;
    u.q = *(const uint4*)(in + o);
#pragma unroll
    for (int j = 0; j < 8; ++j) {
        float x = (float)u.h[j] * sa[c0 + j] + sb[c0 + j];
        u.h[j] = (_Float16)(x > 0.f ? x : 0.f);
    }
    *(uint4*)(out + o) = u.q;
}

// ---------------------------------------------------------------------------
// logits: x2 = relu(y3*a2+b2); out[j] = x2 . c4_w[j] + c4_b[j]
// ---------------------------------------------------------------------------
__global__ __launch_bounds__(256) void logits_kernel(
    const _Float16* __restrict__ y3, const float* __restrict__ alpha,
    const float* __restrict__ beta, const float* __restrict__ c4w,
    const float* __restrict__ c4b, float* __restrict__ out) {
    __shared__ float sw0[128], sw1[128], sa[128], sb[128];
    const int t = threadIdx.x;
    if (t < 128) {
        sw0[t] = c4w[t]; sw1[t] = c4w[128 + t];
        sa[t] = alpha[t]; sb[t] = beta[t];
    }
    __syncthreads();
    const int p = blockIdx.x * 256 + t;
    const _Float16* row = y3 + (size_t)p * 128;
    float a0 = c4b[0], a1 = c4b[1];
    for (int c = 0; c < 128; ++c) {
        float x = (float)row[c] * sa[c] + sb[c];
        x = x > 0.f ? x : 0.f;
        a0 += x * sw0[c];
        a1 += x * sw1[c];
    }
    const int b = p >> 14, q = p & 16383;
    out[((size_t)b * 2 + 0) * 16384 + q] = a0;
    out[((size_t)b * 2 + 1) * 16384 + q] = a1;
}

// ---------------------------------------------------------------------------
extern "C" void kernel_launch(void* const* d_in, const int* in_sizes, int n_in,
                              void* d_out, int out_size, void* d_ws,
                              size_t ws_size, hipStream_t stream) {
    const float* feature  = (const float*)d_in[0];
    const float* vote_xyz = (const float*)d_in[1];
    const float* gu_w  = (const float*)d_in[2];
    const float* gu_b  = (const float*)d_in[3];
    const float* gu_g  = (const float*)d_in[4];
    const float* gu_be = (const float*)d_in[5];
    const float* c1_w  = (const float*)d_in[6];
    const float* c1_b  = (const float*)d_in[7];
    const float* c2_w  = (const float*)d_in[8];
    const float* c2_b  = (const float*)d_in[9];
    const float* bn1_g = (const float*)d_in[10];
    const float* bn1_b = (const float*)d_in[11];
    const float* c3_w  = (const float*)d_in[12];
    const float* c3_b  = (const float*)d_in[13];
    const float* bn2_g = (const float*)d_in[14];
    const float* bn2_b = (const float*)d_in[15];
    const float* c4_w  = (const float*)d_in[16];
    const float* c4_b  = (const float*)d_in[17];

    // output layout: idx(int32) | rn_feature(f32) | logits_0(f32)
    int*   idxOut = (int*)d_out;
    float* rnOut  = (float*)d_out + (size_t)PBN * KNN;
    float* lgOut  = (float*)d_out + (size_t)PBN * KNN + (size_t)BSZ * 128 * NPT;

    // workspace partition
    char* ws = (char*)d_ws;
    size_t off = 0;
    auto alloc = [&](size_t bytes) -> void* {
        void* p = ws + off;
        off = (off + bytes + 255) & ~(size_t)255;
        return p;
    };
    _Float16* featT = (_Float16*)alloc((size_t)PBN * CCH * 2);
    _Float16* rel   = (_Float16*)alloc((size_t)PTOT * CCH * 2);  // reused as x
    _Float16* gu    = (_Float16*)alloc((size_t)PTOT * 256 * 2);  // reused as y3
    _Float16* hn    = (_Float16*)alloc((size_t)PBN * 256 * 2);
    _Float16* y2    = (_Float16*)alloc((size_t)PTOT * 128 * 2);
    _Float16* guW   = (_Float16*)alloc(256 * 128 * 2);
    _Float16* c1W   = (_Float16*)alloc(128 * 256 * 2);
    _Float16* c2W   = (_Float16*)alloc(128 * 128 * 2);
    _Float16* c3W   = (_Float16*)alloc(128 * 128 * 2);
    float* stats    = (float*)alloc((2 * 256 + 2 * 128 + 2 * 128) * 4);
    float* affine   = (float*)alloc((2 * 256 + 2 * 128 + 2 * 128) * 4);

    float* statsGu = stats;
    float* statsY2 = stats + 512;
    float* statsY3 = stats + 768;
    float* alG = affine;        float* beG = affine + 256;
    float* al1 = affine + 512;  float* be1 = affine + 640;
    float* al2 = affine + 768;  float* be2 = affine + 896;
    _Float16* y3 = gu;   // alias: gu consumed before y3 produced
    _Float16* x1 = rel;  // alias: rel consumed before x1 produced

    const float invP = 1.0f / (float)PTOT;

    (void)hipMemsetAsync(stats, 0, (2 * 256 + 2 * 128 + 2 * 128) * 4, stream);

    // weight conversions
    cvt_f16_kernel<<<(256 * 128 + 255) / 256, 256, 0, stream>>>(gu_w, guW, 256 * 128);
    cvt_f16_kernel<<<(128 * 256 + 255) / 256, 256, 0, stream>>>(c1_w, c1W, 128 * 256);
    cvt_f16_kernel<<<(128 * 128 + 255) / 256, 256, 0, stream>>>(c2_w, c2W, 128 * 128);
    cvt_f16_kernel<<<(128 * 128 + 255) / 256, 256, 0, stream>>>(c3_w, c3W, 128 * 128);

    // transpose feature -> featT
    transpose_kernel<<<dim3(NPT / 32, CCH / 32, BSZ), dim3(32, 8), 0, stream>>>(
        feature, featT);

    // KNN -> idx (output 0)
    knn_kernel<<<BSZ * (NPT / 256), 256, 0, stream>>>(vote_xyz, idxOut);

    // rel = feat_i + feat_j  (f16)
    gather_rel_kernel<<<PBN, 128, 0, stream>>>(featT, idxOut, rel);

    // gu = rel @ gu_w^T + gu_b
    gemm_f16_kernel<0><<<dim3(PTOT / 128, 2), 256, 0, stream>>>(
        rel, guW, gu_b, gu, nullptr, 128, 256);
    stats_kernel<<<PTOT / 1024, 256, 0, stream>>>(gu, statsGu, 256, 1024);
    finalize_bn_kernel<<<1, 256, 0, stream>>>(statsGu, gu_g, gu_be, alG, beG,
                                              256, invP);
    // hn = affine(mean_k gu)
    reducek_kernel<<<PBN, 256, 0, stream>>>(gu, alG, beG, hn);
    // rn_feature = hn @ c1_w^T + c1_b  (stored (b,128,N) f32)
    gemm_f16_kernel<1><<<dim3(PBN / 128, 1), 256, 0, stream>>>(
        hn, c1W, c1_b, nullptr, rnOut, 256, 128);

    // y2 = rel @ c2_w^T + c2_b
    gemm_f16_kernel<0><<<dim3(PTOT / 128, 1), 256, 0, stream>>>(
        rel, c2W, c2_b, y2, nullptr, 128, 128);
    stats_kernel<<<PTOT / 1024, 256, 0, stream>>>(y2, statsY2, 128, 1024);
    finalize_bn_kernel<<<1, 256, 0, stream>>>(statsY2, bn1_g, bn1_b, al1, be1,
                                              128, invP);
    // x1 = relu(affine(y2))   (rel is dead now; reuse its buffer)
    affine_relu_kernel<<<PTOT / 16, 256, 0, stream>>>(y2, al1, be1, x1);
    // y3 = x1 @ c3_w^T + c3_b
    gemm_f16_kernel<0><<<dim3(PTOT / 128, 1), 256, 0, stream>>>(
        x1, c3W, c3_b, y3, nullptr, 128, 128);
    stats_kernel<<<PTOT / 1024, 256, 0, stream>>>(y3, statsY3, 128, 1024);
    finalize_bn_kernel<<<1, 256, 0, stream>>>(statsY3, bn2_g, bn2_b, al2, be2,
                                              128, invP);

    // logits = relu(affine(y3)) @ c4_w^T + c4_b  (stored (b,2,N*K) f32)
    logits_kernel<<<PTOT / 256, 256, 0, stream>>>(y3, al2, be2, c4_w, c4_b,
                                                  lgOut);
}
